// STP_GR_Net_31_1202590843141
// MI455X (gfx1250) — matrix operand order repaired and checked by reading, not compiled, not run on hardware
//
#include <hip/hip_runtime.h>
#include <hip/hip_bf16.h>

#define DEVINL __device__ __forceinline__

typedef __attribute__((ext_vector_type(16))) _Float16 v16h;
typedef __attribute__((ext_vector_type(8)))  _Float16 v8h;
typedef __attribute__((ext_vector_type(8)))  float    v8f;

// ---------------------------------------------------------------------------
// WMMA helpers (CDNA5 wave32, v_wmma_f32_16x16x32_f16)
// ---------------------------------------------------------------------------
DEVINL v8f wmma_f16(v16h a, v16h b, v8f c) {
  return __builtin_amdgcn_wmma_f32_16x16x32_f16(
      /*neg_a=*/false, a, /*neg_b=*/false, b,
      /*c_mod=*/(short)0, c, /*reuse_a=*/false, /*reuse_b=*/false);
}

union AFrag { v16h v; v8h h[2]; };

// A-matrix fragment (16x32 f16) from row-major f16 buffer (row_stride halves).
// Lane L: row = L&15, g = L>>4. halves 0..7 <-> K = kb+8g.. ; 8..15 <-> K+16.
DEVINL v16h load_a(const _Float16* base, int row_stride, int kb, int lane) {
  int row = lane & 15;
  int g   = (lane >> 4) & 1;
  const _Float16* p = base + row * row_stride + kb + 8 * g;
  AFrag af;
  af.h[0] = *(const v8h*)(p);
  af.h[1] = *(const v8h*)(p + 16);
  return af.v;
}

// B-matrix fragment from pre-swizzled fragment buffer: frag = 32 lanes x 16 halves.
DEVINL v16h load_b(const _Float16* fragbase, int fragidx, int lane) {
  return *(const v16h*)(fragbase + ((size_t)fragidx << 9) + lane * 16);
}

DEVINL v8f bias_c(const float* __restrict__ b, int ntile, int lane) {
  float bv = b[ntile * 16 + (lane & 15)];
  v8f c;
#pragma unroll
  for (int r = 0; r < 8; ++r) c[r] = bv;
  return c;
}
DEVINL v8f bias2_c(const float* __restrict__ b1, const float* __restrict__ b2,
                   int ntile, int lane) {
  int ch = ntile * 16 + (lane & 15);
  float bv = b1[ch] + b2[ch];
  v8f c;
#pragma unroll
  for (int r = 0; r < 8; ++r) c[r] = bv;
  return c;
}

// Fast activations: v_exp_f32 / v_log_f32 / v_rcp_f32, no precise-div chains.
DEVINL float frcp(float x)      { return __builtin_amdgcn_rcpf(x); }
DEVINL float sigm(float x)      { return frcp(1.f + __expf(-x)); }
DEVINL float tanh_fast(float x) { return 1.f - 2.f * frcp(__expf(2.f * x) + 1.f); }
DEVINL float lrelu(float x)     { return x >= 0.f ? x : 0.1f * x; }
DEVINL float splus(float x)     { return x > 20.f ? x : __logf(1.f + __expf(x)); }

// ---------------------------------------------------------------------------
// Weight fragment pre-pack: W[out][in] f32 -> f16 B-fragments, zero padded
// in both K (in_dim) and N (out_dim).
// ---------------------------------------------------------------------------
__global__ void k_pack(const float* __restrict__ W, _Float16* __restrict__ dst,
                       int Cc, int in_dim, int out_dim, int total) {
  int tid = blockIdx.x * 256 + threadIdx.x;
  if (tid >= total) return;
  int f      = tid >> 9;
  int within = tid & 511;
  int lane   = within >> 4;
  int h      = within & 15;
  int j = f / Cc, c = f - j * Cc;
  int n = lane & 15, g = lane >> 4;
  int kidx = 32 * c + ((h < 8) ? (h + 8 * g) : (h + 8 + 8 * g));
  int nidx = j * 16 + n;
  float v = (kidx < in_dim && nidx < out_dim)
                ? W[(size_t)nidx * in_dim + kidx] : 0.f;
  dst[tid] = (_Float16)v;
}

__global__ void k_zero(float* __restrict__ p, int n) {
  int i = blockIdx.x * 256 + threadIdx.x;
  if (i < n) p[i] = 0.f;
}

// ---------------------------------------------------------------------------
// Encoder: embed + 16-step GRU(32->64) + lrelu + W_dyn + lrelu.
// One wave per 16 nodes. r/z gates accumulate gi+gh into a single WMMA C.
// f32 hidden state lives in LDS (C-fragment layout) -> no movrels register
// array indexing; f16 copy in LDS row-major for A-fragment reloads.
// ---------------------------------------------------------------------------
__global__ void k_encoder(const float* __restrict__ x,
                          const float* __restrict__ Wip, const float* __restrict__ bip,
                          const _Float16* __restrict__ fgWih,
                          const _Float16* __restrict__ fgWhh,
                          const _Float16* __restrict__ fgWdyn,
                          const float* __restrict__ gbih, const float* __restrict__ gbhh,
                          const float* __restrict__ bdyn,
                          float* __restrict__ hist32, _Float16* __restrict__ hist16,
                          int ntiles) {
  __shared__ __align__(16) _Float16 Hs[8][16 * 72];
  __shared__ __align__(32) float    CSs[8][4 * 32 * 8];
  int lane = threadIdx.x & 31;
  int wv   = threadIdx.x >> 5;
  int wt   = blockIdx.x * 8 + wv;
  if (wt >= ntiles) return;
  _Float16* Hb = &Hs[wv][0];
  float*    CS = &CSs[wv][0];
  int node0 = wt * 16;
  int row = lane & 15, g = (lane >> 4) & 1;

  // per-lane embedding constants (the 16 K-channels this lane packs)
  float w0[16], w1[16], bb[16];
#pragma unroll
  for (int h = 0; h < 16; ++h) {
    int k = (h < 8) ? (h + 8 * g) : (h + 8 + 8 * g);
    w0[h] = Wip[k * 2 + 0];
    w1[h] = Wip[k * 2 + 1];
    bb[h] = bip[k];
  }

  for (int i = lane; i < 16 * 72; i += 32) Hb[i] = (_Float16)0.f;
  {
    v8f z;
#pragma unroll
    for (int r = 0; r < 8; ++r) z[r] = 0.f;
#pragma unroll
    for (int j = 0; j < 4; ++j) *(v8f*)&CS[(j * 32 + lane) * 8] = z;
  }

#pragma clang loop unroll(disable)
  for (int t = 0; t < 16; ++t) {
    float x0 = x[((size_t)(node0 + row) * 16 + t) * 2 + 0];
    float x1 = x[((size_t)(node0 + row) * 16 + t) * 2 + 1];
    v16h ea;
#pragma unroll
    for (int h = 0; h < 16; ++h)
      ea[h] = (_Float16)lrelu(x0 * w0[h] + x1 * w1[h] + bb[h]);

    v16h ha0 = load_a(Hb, 72, 0, lane);
    v16h ha1 = load_a(Hb, 72, 32, lane);
#pragma clang loop unroll(disable)
    for (int j = 0; j < 4; ++j) {
      // r-gate sum: ea@Wih_r + h@Whh_r + (bih_r + bhh_r)  -- one accumulator
      v8f sr = bias2_c(gbih, gbhh, j, lane);
      sr = wmma_f16(ea,  load_b(fgWih, j,           lane), sr);
      sr = wmma_f16(ha0, load_b(fgWhh, j * 2,       lane), sr);
      sr = wmma_f16(ha1, load_b(fgWhh, j * 2 + 1,   lane), sr);
      // z-gate sum
      v8f sz = bias2_c(gbih, gbhh, j + 4, lane);
      sz = wmma_f16(ea,  load_b(fgWih, j + 4,           lane), sz);
      sz = wmma_f16(ha0, load_b(fgWhh, (j + 4) * 2,     lane), sz);
      sz = wmma_f16(ha1, load_b(fgWhh, (j + 4) * 2 + 1, lane), sz);
      // n-gate needs i_n and h_n separately (n = tanh(i_n + r*h_n))
      v8f in_ = bias_c(gbih, j + 8, lane);
      in_ = wmma_f16(ea, load_b(fgWih, j + 8, lane), in_);
      v8f hn = bias_c(gbhh, j + 8, lane);
      hn = wmma_f16(ha0, load_b(fgWhh, (j + 8) * 2,     lane), hn);
      hn = wmma_f16(ha1, load_b(fgWhh, (j + 8) * 2 + 1, lane), hn);

      v8f cc = *(v8f*)&CS[(j * 32 + lane) * 8];
#pragma unroll
      for (int r = 0; r < 8; ++r) {
        float rr = sigm(sr[r]);
        float zz = sigm(sz[r]);
        float nn = tanh_fast(in_[r] + rr * hn[r]);
        float hv = (1.f - zz) * nn + zz * cc[r];
        cc[r] = hv;
        Hb[(r + 8 * g) * 72 + j * 16 + (lane & 15)] = (_Float16)hv;
      }
      *(v8f*)&CS[(j * 32 + lane) * 8] = cc;
    }
  }

  // hist = lrelu( lrelu(hT) @ Wdyn^T + bdyn )
#pragma clang loop unroll(disable)
  for (int j = 0; j < 4; ++j) {
    v8f cc = *(v8f*)&CS[(j * 32 + lane) * 8];
#pragma unroll
    for (int r = 0; r < 8; ++r)
      Hb[(r + 8 * g) * 72 + j * 16 + (lane & 15)] = (_Float16)lrelu(cc[r]);
  }
  v16h da0 = load_a(Hb, 72, 0, lane);
  v16h da1 = load_a(Hb, 72, 32, lane);
#pragma clang loop unroll(disable)
  for (int j = 0; j < 4; ++j) {
    v8f c = bias_c(bdyn, j, lane);
    c = wmma_f16(da0, load_b(fgWdyn, j * 2,     lane), c);
    c = wmma_f16(da1, load_b(fgWdyn, j * 2 + 1, lane), c);
#pragma unroll
    for (int r = 0; r < 8; ++r) {
      float v = lrelu(c[r]);
      size_t idx = (size_t)(node0 + r + 8 * g) * 64 + j * 16 + (lane & 15);
      hist32[idx] = v;
      hist16[idx] = (_Float16)v;
    }
  }
}

// ---------------------------------------------------------------------------
// CGConv: one wave per 16 edges. Gather to z[16][160] (K padded 130->160),
// 4 N-tiles x (Wf,Ws) x 5 K-chunks WMMA, msg = sigmoid*softplus, atomic
// scatter-add into L2-resident agg.
// ---------------------------------------------------------------------------
__global__ void k_cgconv(const _Float16* __restrict__ xin16,
                         const int* __restrict__ srcp, const int* __restrict__ dstp,
                         const float* __restrict__ eattr,
                         const _Float16* __restrict__ fgWf,
                         const _Float16* __restrict__ fgWs,
                         const float* __restrict__ bf, const float* __restrict__ bs,
                         float* __restrict__ agg, int etiles) {
  __shared__ __align__(16) _Float16 Zs[8][16 * 160];
  int lane = threadIdx.x & 31, wv = threadIdx.x >> 5;
  int wt = blockIdx.x * 8 + wv;
  if (wt >= etiles) return;
  _Float16* Z = &Zs[wv][0];
  int e0 = wt * 16;
  int row = lane & 15, g = (lane >> 4) & 1;
  int e  = e0 + row;
  int de = dstp[e], se = srcp[e];
  __builtin_prefetch(xin16 + (size_t)se * 64, 0, 1);

  {
    const v8h* p = (const v8h*)(xin16 + (size_t)de * 64 + g * 32);
    v8h a = p[0], b = p[1], c = p[2], d = p[3];
    v8h* q = (v8h*)(Z + row * 160 + g * 32);
    q[0] = a; q[1] = b; q[2] = c; q[3] = d;
  }
  {
    const v8h* p = (const v8h*)(xin16 + (size_t)se * 64 + g * 32);
    v8h a = p[0], b = p[1], c = p[2], d = p[3];
    v8h* q = (v8h*)(Z + row * 160 + 64 + g * 32);
    q[0] = a; q[1] = b; q[2] = c; q[3] = d;
  }
  if (g == 0) {
    Z[row * 160 + 128] = (_Float16)eattr[(size_t)e * 2 + 0];
    Z[row * 160 + 129] = (_Float16)eattr[(size_t)e * 2 + 1];
    for (int k = 130; k < 144; ++k) Z[row * 160 + k] = (_Float16)0.f;
  } else {
    for (int k = 144; k < 160; ++k) Z[row * 160 + k] = (_Float16)0.f;
  }

#pragma clang loop unroll(disable)
  for (int j = 0; j < 4; ++j) {
    v8f cf = bias_c(bf, j, lane);
    v8f cs = bias_c(bs, j, lane);
#pragma unroll
    for (int c = 0; c < 5; ++c) {
      v16h a = load_a(Z, 160, 32 * c, lane);
      cf = wmma_f16(a, load_b(fgWf, j * 5 + c, lane), cf);
      cs = wmma_f16(a, load_b(fgWs, j * 5 + c, lane), cs);
    }
#pragma unroll
    for (int r = 0; r < 8; ++r) {
      float m = sigm(cf[r]) * splus(cs[r]);
      int dd = dstp[e0 + r + 8 * g];
      atomicAdd(&agg[(size_t)dd * 64 + j * 16 + (lane & 15)], m);
    }
  }
}

// BN (inference stats) + residual; emits f16 copy; re-zeros agg for replay.
__global__ void k_bnres(const float* __restrict__ xin32, const float* __restrict__ aggin,
                        const float* __restrict__ gamma, const float* __restrict__ beta,
                        const float* __restrict__ mean,  const float* __restrict__ var,
                        float* __restrict__ out32, _Float16* __restrict__ out16,
                        float* __restrict__ agg_reset, int total) {
  int i = blockIdx.x * 256 + threadIdx.x;
  if (i >= total) return;
  int ch = i & 63;
  float v = xin32[i] +
            (aggin[i] - mean[ch]) * rsqrtf(var[ch] + 1e-5f) * gamma[ch] + beta[ch];
  if (out32) out32[i] = v;
  out16[i] = (_Float16)v;
  agg_reset[i] = 0.f;
}

// ---------------------------------------------------------------------------
// Target stage: tar = lrelu(f2[tgt]@Wnbrs^T+b); enc = [hist[tgt], tar];
// gi0 = enc @ lstm0_Wih^T + bih0 + bhh0 (time-invariant layer-0 input GEMM),
// stored pre-swizzled in C-fragment layout.
// ---------------------------------------------------------------------------
__global__ void k_target(const int* __restrict__ tgt,
                         const _Float16* __restrict__ hist16,
                         const _Float16* __restrict__ f16,
                         const _Float16* __restrict__ fgWnbrs, const float* __restrict__ bnbrs,
                         const _Float16* __restrict__ fgWih0,
                         const float* __restrict__ bih0, const float* __restrict__ bhh0,
                         float* __restrict__ gi0out, int btiles) {
  __shared__ __align__(16) _Float16 Es[4][16 * 128];
  __shared__ __align__(16) _Float16 Ts[4][16 * 64];
  int lane = threadIdx.x & 31, wv = threadIdx.x >> 5;
  int wt = blockIdx.x * 4 + wv;
  if (wt >= btiles) return;
  _Float16* Eb = &Es[wv][0];
  _Float16* Tb = &Ts[wv][0];
  int row = lane & 15, g = (lane >> 4) & 1;
  int tg = tgt[wt * 16 + row];
  {
    const v8h* p = (const v8h*)(hist16 + (size_t)tg * 64 + g * 32);
    v8h a = p[0], b = p[1], c = p[2], d = p[3];
    v8h* q = (v8h*)(Eb + row * 128 + g * 32);
    q[0] = a; q[1] = b; q[2] = c; q[3] = d;
  }
  {
    const v8h* p = (const v8h*)(f16 + (size_t)tg * 64 + g * 32);
    v8h a = p[0], b = p[1], c = p[2], d = p[3];
    v8h* q = (v8h*)(Tb + row * 64 + g * 32);
    q[0] = a; q[1] = b; q[2] = c; q[3] = d;
  }
  v16h a0 = load_a(Tb, 64, 0, lane);
  v16h a1 = load_a(Tb, 64, 32, lane);
#pragma clang loop unroll(disable)
  for (int j = 0; j < 4; ++j) {
    v8f c = bias_c(bnbrs, j, lane);
    c = wmma_f16(a0, load_b(fgWnbrs, j * 2,     lane), c);
    c = wmma_f16(a1, load_b(fgWnbrs, j * 2 + 1, lane), c);
#pragma unroll
    for (int r = 0; r < 8; ++r)
      Eb[(r + 8 * g) * 128 + 64 + j * 16 + (lane & 15)] = (_Float16)lrelu(c[r]);
  }
#pragma clang loop unroll(disable)
  for (int j = 0; j < 32; ++j) {
    v8f c = bias2_c(bih0, bhh0, j, lane);
#pragma unroll
    for (int cc = 0; cc < 4; ++cc) {
      v16h a = load_a(Eb, 128, 32 * cc, lane);
      c = wmma_f16(a, load_b(fgWih0, j * 4 + cc, lane), c);
    }
    *(v8f*)(gi0out + (((size_t)wt * 32 + j) * 32 + lane) * 8) = c;
  }
}

// ---------------------------------------------------------------------------
// LSTM decoder: ONE wave per block, 16 targets, 25 steps, 2 layers DEC=128.
// Cell state lives in LDS in C-fragment layout (no register arrays/spills);
// h via double-buffered LDS f16 for A-fragment reloads. Output projection is
// a padded 16-wide WMMA N-tile (2 valid columns).
// ---------------------------------------------------------------------------
__global__ void k_lstm(const float* __restrict__ gi0,
                       const _Float16* __restrict__ fgWhh0,
                       const _Float16* __restrict__ fgWih1,
                       const _Float16* __restrict__ fgWhh1,
                       const float* __restrict__ bih1, const float* __restrict__ bhh1,
                       const _Float16* __restrict__ fgWop, const float* __restrict__ bop,
                       float* __restrict__ out, int btiles) {
  __shared__ __align__(16) _Float16 H0s[2][16 * 128];
  __shared__ __align__(16) _Float16 H1s[2][16 * 128];
  __shared__ __align__(32) float    CS0[8 * 32 * 8];
  __shared__ __align__(32) float    CS1[8 * 32 * 8];
  int lane = threadIdx.x & 31;
  int wt = blockIdx.x;
  if (wt >= btiles) return;
  int g = (lane >> 4) & 1;
  for (int i = lane; i < 16 * 128; i += 32) {
    H0s[0][i] = (_Float16)0.f; H0s[1][i] = (_Float16)0.f;
    H1s[0][i] = (_Float16)0.f; H1s[1][i] = (_Float16)0.f;
    CS0[i] = 0.f; CS1[i] = 0.f;
  }
  const float* gi0w = gi0 + (size_t)wt * 8192;
  int cur = 0;

#pragma clang loop unroll(disable)
  for (int t = 0; t < 25; ++t) {
    _Float16* H0r = H0s[cur];
    _Float16* H1r = H1s[cur];
    _Float16* H0w = H0s[cur ^ 1];
    _Float16* H1w = H1s[cur ^ 1];

    // ---- layer 0: gates = gi0(const, pre-swizzled) + h0 @ Whh0^T ----
#pragma clang loop unroll(disable)
    for (int j = 0; j < 8; ++j) {
      v8f gi_ = *(const v8f*)(gi0w + ((size_t)(0 * 8 + j) * 32 + lane) * 8);
      v8f gf_ = *(const v8f*)(gi0w + ((size_t)(1 * 8 + j) * 32 + lane) * 8);
      v8f gg_ = *(const v8f*)(gi0w + ((size_t)(2 * 8 + j) * 32 + lane) * 8);
      v8f go_ = *(const v8f*)(gi0w + ((size_t)(3 * 8 + j) * 32 + lane) * 8);
#pragma unroll
      for (int c = 0; c < 4; ++c) {
        v16h a = load_a(H0r, 128, 32 * c, lane);
        gi_ = wmma_f16(a, load_b(fgWhh0, (0 * 8 + j) * 4 + c, lane), gi_);
        gf_ = wmma_f16(a, load_b(fgWhh0, (1 * 8 + j) * 4 + c, lane), gf_);
        gg_ = wmma_f16(a, load_b(fgWhh0, (2 * 8 + j) * 4 + c, lane), gg_);
        go_ = wmma_f16(a, load_b(fgWhh0, (3 * 8 + j) * 4 + c, lane), go_);
      }
      v8f cc = *(v8f*)&CS0[(j * 32 + lane) * 8];
#pragma unroll
      for (int r = 0; r < 8; ++r) {
        float c2 = sigm(gf_[r]) * cc[r] + sigm(gi_[r]) * tanh_fast(gg_[r]);
        cc[r] = c2;
        H0w[(r + 8 * g) * 128 + j * 16 + (lane & 15)] =
            (_Float16)(sigm(go_[r]) * tanh_fast(c2));
      }
      *(v8f*)&CS0[(j * 32 + lane) * 8] = cc;
    }

    // ---- layer 1: gates = h0_t @ Wih1^T + h1 @ Whh1^T + (bih1+bhh1) ----
#pragma clang loop unroll(disable)
    for (int j = 0; j < 8; ++j) {
      v8f gi_ = bias2_c(bih1, bhh1, 0 * 8 + j, lane);
      v8f gf_ = bias2_c(bih1, bhh1, 1 * 8 + j, lane);
      v8f gg_ = bias2_c(bih1, bhh1, 2 * 8 + j, lane);
      v8f go_ = bias2_c(bih1, bhh1, 3 * 8 + j, lane);
#pragma unroll
      for (int c = 0; c < 4; ++c) {
        v16h a = load_a(H0w, 128, 32 * c, lane);
        gi_ = wmma_f16(a, load_b(fgWih1, (0 * 8 + j) * 4 + c, lane), gi_);
        gf_ = wmma_f16(a, load_b(fgWih1, (1 * 8 + j) * 4 + c, lane), gf_);
        gg_ = wmma_f16(a, load_b(fgWih1, (2 * 8 + j) * 4 + c, lane), gg_);
        go_ = wmma_f16(a, load_b(fgWih1, (3 * 8 + j) * 4 + c, lane), go_);
        v16h b = load_a(H1r, 128, 32 * c, lane);
        gi_ = wmma_f16(b, load_b(fgWhh1, (0 * 8 + j) * 4 + c, lane), gi_);
        gf_ = wmma_f16(b, load_b(fgWhh1, (1 * 8 + j) * 4 + c, lane), gf_);
        gg_ = wmma_f16(b, load_b(fgWhh1, (2 * 8 + j) * 4 + c, lane), gg_);
        go_ = wmma_f16(b, load_b(fgWhh1, (3 * 8 + j) * 4 + c, lane), go_);
      }
      v8f cc = *(v8f*)&CS1[(j * 32 + lane) * 8];
#pragma unroll
      for (int r = 0; r < 8; ++r) {
        float c2 = sigm(gf_[r]) * cc[r] + sigm(gi_[r]) * tanh_fast(gg_[r]);
        cc[r] = c2;
        H1w[(r + 8 * g) * 128 + j * 16 + (lane & 15)] =
            (_Float16)(sigm(go_[r]) * tanh_fast(c2));
      }
      *(v8f*)&CS1[(j * 32 + lane) * 8] = cc;
    }

    // ---- output projection: h1 @ Wop^T (2 valid cols in a padded 16-tile) ----
    v8f oc;
#pragma unroll
    for (int r = 0; r < 8; ++r) oc[r] = 0.f;
#pragma unroll
    for (int c = 0; c < 4; ++c) {
      v16h a = load_a(H1w, 128, 32 * c, lane);
      oc = wmma_f16(a, load_b(fgWop, c, lane), oc);
    }
    int n = lane & 15;
    if (n < 2) {
#pragma unroll
      for (int r = 0; r < 8; ++r)
        out[((size_t)(wt * 16 + r + 8 * g)) * 50 + t * 2 + n] = oc[r] + bop[n];
    }
    cur ^= 1;
  }
}

// ---------------------------------------------------------------------------
// Host-side orchestration
// ---------------------------------------------------------------------------
extern "C" void kernel_launch(void* const* d_in, const int* in_sizes, int n_in,
                              void* d_out, int out_size, void* d_ws, size_t ws_size,
                              hipStream_t stream) {
  (void)n_in; (void)out_size; (void)ws_size;
  const float* x     = (const float*)d_in[0];
  const int*   eidx  = (const int*)  d_in[1];
  const float* eattr = (const float*)d_in[2];
  const int*   tgt   = (const int*)  d_in[3];
  const float* Wip   = (const float*)d_in[4];
  const float* bip   = (const float*)d_in[5];
  const float* gWih  = (const float*)d_in[6];
  const float* gWhh  = (const float*)d_in[7];
  const float* gbih  = (const float*)d_in[8];
  const float* gbhh  = (const float*)d_in[9];
  const float* Wdyn  = (const float*)d_in[10];
  const float* bdyn  = (const float*)d_in[11];
  const float* cg1Wf = (const float*)d_in[12];
  const float* cg1bf = (const float*)d_in[13];
  const float* cg1Ws = (const float*)d_in[14];
  const float* cg1bs = (const float*)d_in[15];
  const float* cg1g  = (const float*)d_in[16];
  const float* cg1b  = (const float*)d_in[17];
  const float* cg1m  = (const float*)d_in[18];
  const float* cg1v  = (const float*)d_in[19];
  const float* cg2Wf = (const float*)d_in[20];
  const float* cg2bf = (const float*)d_in[21];
  const float* cg2Ws = (const float*)d_in[22];
  const float* cg2bs = (const float*)d_in[23];
  const float* cg2g  = (const float*)d_in[24];
  const float* cg2b  = (const float*)d_in[25];
  const float* cg2m  = (const float*)d_in[26];
  const float* cg2v  = (const float*)d_in[27];
  const float* Wnbrs = (const float*)d_in[28];
  const float* bnbrs = (const float*)d_in[29];
  const float* l0Wih = (const float*)d_in[30];
  const float* l0Whh = (const float*)d_in[31];
  const float* l0bih = (const float*)d_in[32];
  const float* l0bhh = (const float*)d_in[33];
  const float* l1Wih = (const float*)d_in[34];
  const float* l1Whh = (const float*)d_in[35];
  const float* l1bih = (const float*)d_in[36];
  const float* l1bhh = (const float*)d_in[37];
  const float* Wop   = (const float*)d_in[38];
  const float* bop   = (const float*)d_in[39];

  const int N = in_sizes[0] / 32;   // N*T*2 / (16*2)
  const int E = in_sizes[1] / 2;
  const int B = in_sizes[3];
  const int* srcp = eidx;
  const int* dstp = eidx + E;

  char* base = (char*)d_ws;
  size_t off = 0;
  auto carve = [&](size_t bytes) -> char* {
    char* p = base + off;
    off = (off + bytes + 255) & ~(size_t)255;
    return p;
  };
  const size_t FR = 1024;  // bytes per fragment (512 halves)
  _Float16* fg_gWih  = (_Float16*)carve(12  * FR);  // [192x32]  J=12 C=1
  _Float16* fg_gWhh  = (_Float16*)carve(24  * FR);  // [192x64]  J=12 C=2
  _Float16* fg_Wdyn  = (_Float16*)carve(8   * FR);  // [64x64]   J=4  C=2
  _Float16* fg_c1Wf  = (_Float16*)carve(20  * FR);  // [64x130]  J=4  C=5
  _Float16* fg_c1Ws  = (_Float16*)carve(20  * FR);
  _Float16* fg_c2Wf  = (_Float16*)carve(20  * FR);
  _Float16* fg_c2Ws  = (_Float16*)carve(20  * FR);
  _Float16* fg_Wnbrs = (_Float16*)carve(8   * FR);  // [64x64]
  _Float16* fg_l0Wih = (_Float16*)carve(128 * FR);  // [512x128] J=32 C=4
  _Float16* fg_l0Whh = (_Float16*)carve(128 * FR);
  _Float16* fg_l1Wih = (_Float16*)carve(128 * FR);
  _Float16* fg_l1Whh = (_Float16*)carve(128 * FR);
  _Float16* fg_Wop   = (_Float16*)carve(4   * FR);  // [2x128] padded, J=1 C=4
  float*     hist32 = (float*)    carve((size_t)N * 64 * 4);
  _Float16*  hist16 = (_Float16*) carve((size_t)N * 64 * 2);
  float*     f1_32  = (float*)    carve((size_t)N * 64 * 4);
  _Float16*  f1_16  = (_Float16*) carve((size_t)N * 64 * 2);
  _Float16*  f2_16  = (_Float16*) carve((size_t)N * 64 * 2);
  float*     agg    = (float*)    carve((size_t)N * 64 * 4);
  float*     gi0    = (float*)    carve((size_t)B * 512 * 4);

  auto pack = [&](const float* W, _Float16* dst, int J, int C,
                  int in_dim, int out_dim) {
    int total = J * C * 512;
    k_pack<<<(total + 255) / 256, 256, 0, stream>>>(W, dst, C, in_dim, out_dim, total);
  };
  pack(gWih,  fg_gWih,  12, 1, 32,  192);
  pack(gWhh,  fg_gWhh,  12, 2, 64,  192);
  pack(Wdyn,  fg_Wdyn,   4, 2, 64,  64);
  pack(cg1Wf, fg_c1Wf,   4, 5, 130, 64);
  pack(cg1Ws, fg_c1Ws,   4, 5, 130, 64);
  pack(cg2Wf, fg_c2Wf,   4, 5, 130, 64);
  pack(cg2Ws, fg_c2Ws,   4, 5, 130, 64);
  pack(Wnbrs, fg_Wnbrs,  4, 2, 64,  64);
  pack(l0Wih, fg_l0Wih, 32, 4, 128, 512);
  pack(l0Whh, fg_l0Whh, 32, 4, 128, 512);
  pack(l1Wih, fg_l1Wih, 32, 4, 128, 512);
  pack(l1Whh, fg_l1Whh, 32, 4, 128, 512);
  pack(Wop,   fg_Wop,    1, 4, 128, 2);

  int total64 = N * 64;
  k_zero<<<(total64 + 255) / 256, 256, 0, stream>>>(agg, total64);

  int ntiles = N / 16;
  k_encoder<<<(ntiles + 7) / 8, 256, 0, stream>>>(
      x, Wip, bip, fg_gWih, fg_gWhh, fg_Wdyn, gbih, gbhh, bdyn,
      hist32, hist16, ntiles);

  int etiles = E / 16;
  k_cgconv<<<(etiles + 7) / 8, 256, 0, stream>>>(
      hist16, srcp, dstp, eattr, fg_c1Wf, fg_c1Ws, cg1bf, cg1bs, agg, etiles);
  k_bnres<<<(total64 + 255) / 256, 256, 0, stream>>>(
      hist32, agg, cg1g, cg1b, cg1m, cg1v, f1_32, f1_16, agg, total64);

  k_cgconv<<<(etiles + 7) / 8, 256, 0, stream>>>(
      f1_16, srcp, dstp, eattr, fg_c2Wf, fg_c2Ws, cg2bf, cg2bs, agg, etiles);
  k_bnres<<<(total64 + 255) / 256, 256, 0, stream>>>(
      f1_32, agg, cg2g, cg2b, cg2m, cg2v, (float*)nullptr, f2_16, agg, total64);

  int btiles = B / 16;
  k_target<<<(btiles + 3) / 4, 128, 0, stream>>>(
      tgt, hist16, f2_16, fg_Wnbrs, bnbrs, fg_l0Wih, l0bih, l0bhh, gi0, btiles);

  k_lstm<<<btiles, 32, 0, stream>>>(
      gi0, fg_l0Whh, fg_l1Wih, fg_l1Whh, l1bih, l1bhh, fg_Wop, bop,
      (float*)d_out, btiles);
}
